// Upsample_69990787055779
// MI455X (gfx1250) — matrix operand order repaired and checked by reading, not compiled, not run on hardware
//
#include <hip/hip_runtime.h>

// ---------------------------------------------------------------------------
// Shapes (fixed by the reference)
// ---------------------------------------------------------------------------
#define BATCH 2
#define SEQ   512
#define SUP   1024          // upsampled seq
#define HID   2048
#define NHEAD 32
#define HDIM  64
#define TKV   1536          // SEQ + SUP
#define INTER 5632

typedef __attribute__((ext_vector_type(16))) __bf16 v16bf;
typedef __attribute__((ext_vector_type(8)))  float  v8f;

union Frag { unsigned u[8]; v16bf v; };

__device__ __forceinline__ unsigned short f2bf(float f) {
    unsigned u = __float_as_uint(f);
    u += 0x7FFFu + ((u >> 16) & 1u);          // round to nearest even
    return (unsigned short)(u >> 16);
}

__device__ __forceinline__ v8f wmma_bf16(v16bf a, v16bf b, v8f c) {
    return __builtin_amdgcn_wmma_f32_16x16x32_bf16(
        false, a, false, b, (short)0, c, false, false);
}

// A-fragment K offset for VGPR r (pairs of K are contiguous): K = 16*(r/4) + 8*g + 2*(r%4)
__device__ __forceinline__ int kOffA(int r, int g) {
    return ((r >> 2) << 4) | (g << 3) | ((r & 3) << 1);
}
// B-fragment K offset (same as A): K = 16*(r/4) + 8*g + 2*(r%4)
__device__ __forceinline__ int kOffB(int r, int g) {
    return ((r >> 2) << 4) | (g << 3) | ((r & 3) << 1);
}

// ---------------------------------------------------------------------------
// fp32 -> bf16 conversion kernels
// ---------------------------------------------------------------------------
__global__ void cvt_kernel(const float* __restrict__ in,
                           unsigned short* __restrict__ out, long n) {
    long i = (long)blockIdx.x * blockDim.x + threadIdx.x;
    if (i < n) out[i] = f2bf(in[i]);
}

__global__ void cvt_batched_kernel(const float* __restrict__ in, long sIn,
                                   unsigned short* __restrict__ out, long sOut,
                                   long nPer) {
    long b = blockIdx.y;
    long i = (long)blockIdx.x * blockDim.x + threadIdx.x;
    if (i < nPer) out[b * sOut + i] = f2bf(in[b * sIn + i]);
}

// transpose fp32 [rows x cols] -> bf16 [cols x rows], per batch
__global__ void transpose_cvt_kernel(const float* __restrict__ in,
                                     unsigned short* __restrict__ out,
                                     int rows, int cols, long sIn, long sOut) {
    __shared__ float t[32][33];
    const int b  = blockIdx.z;
    const float* ip = in + (long)b * sIn;
    unsigned short* op = out + (long)b * sOut;
    const int c0 = blockIdx.x * 32;
    const int r0 = blockIdx.y * 32;
    const int tx = threadIdx.x, ty = threadIdx.y;   // (32, 8)
#pragma unroll
    for (int i = 0; i < 4; ++i)
        t[ty + i * 8][tx] = ip[(long)(r0 + ty + i * 8) * cols + c0 + tx];
    __syncthreads();
#pragma unroll
    for (int i = 0; i < 4; ++i)
        op[(long)(c0 + ty + i * 8) * rows + r0 + tx] = f2bf(t[tx][ty + i * 8]);
}

// ---------------------------------------------------------------------------
// RMSNorm: one block per row, outputs fp32 (optional) and bf16
// ---------------------------------------------------------------------------
__global__ __launch_bounds__(256)
void rmsnorm_kernel(const float* __restrict__ X, const float* __restrict__ W,
                    float* __restrict__ OF, unsigned short* __restrict__ OB,
                    int H) {
    const long row = blockIdx.x;
    const float* x = X + row * (long)H;
    __shared__ float red[256];
    float s = 0.f;
    for (int i = threadIdx.x; i < H; i += 256) { float v = x[i]; s += v * v; }
    red[threadIdx.x] = s;
    __syncthreads();
    for (int off = 128; off; off >>= 1) {
        if (threadIdx.x < off) red[threadIdx.x] += red[threadIdx.x + off];
        __syncthreads();
    }
    const float rstd = rsqrtf(red[0] / (float)H + 1e-6f);
    for (int i = threadIdx.x; i < H; i += 256) {
        float v = x[i] * rstd * W[i];
        if (OF) OF[row * (long)H + i] = v;
        OB[row * (long)H + i] = f2bf(v);
    }
}

// ---------------------------------------------------------------------------
// silu(g) * u -> bf16
// ---------------------------------------------------------------------------
__global__ void silu_mul_kernel(const float* __restrict__ g,
                                const float* __restrict__ u,
                                unsigned short* __restrict__ out, long n) {
    long i = (long)blockIdx.x * blockDim.x + threadIdx.x;
    if (i < n) {
        float x = g[i];
        out[i] = f2bf(x / (1.f + __expf(-x)) * u[i]);
    }
}

// ---------------------------------------------------------------------------
// Tiled WMMA GEMM:  C[M,N] = A[M,K] * B[N,K]^T  (both operands K-contiguous)
// Block tile 128x64, BK=64, 256 threads = 8 waves, each wave 32x32 (2x2 WMMA).
// Tiles staged with GLOBAL_LOAD_ASYNC_TO_LDS_B128 (double buffered, ASYNCcnt).
// Epilogue: optional per-M bias, fp32 residual add, fp32 store, bf16 store,
// bf16 transposed store.  All dims must divide the tile sizes.
// ---------------------------------------------------------------------------
__global__ __launch_bounds__(256)
void gemm_bf16_kernel(const unsigned short* __restrict__ A, long lda, long sA,
                      const unsigned short* __restrict__ B, long ldb, long sB,
                      float* __restrict__ C, long ldc, long sC,
                      const float* __restrict__ R, long ldr, long sR,
                      const float* __restrict__ bias,
                      unsigned short* __restrict__ OB, long ldo, long sO,
                      unsigned short* __restrict__ OT, long ldt, long sT,
                      int K) {
    constexpr int BM = 128, BN = 64, BK = 64, LST = 72;   // LDS row stride (bf16)
    __shared__ __align__(16) unsigned short As[2][BM * LST];
    __shared__ __align__(16) unsigned short Bs[2][BN * LST];

    const int tid  = threadIdx.x;
    const int wave = tid >> 5, lane = tid & 31;
    const int g = lane >> 4, l16 = lane & 15;
    const int wm = (wave & 3) * 32;       // wave M offset in block tile
    const int wn = (wave >> 2) * 32;      // wave N offset in block tile
    const long m0 = (long)blockIdx.y * BM;
    const long n0 = (long)blockIdx.x * BN;
    const long bz = blockIdx.z;
    A += bz * sA;
    B += bz * sB;

    int kA[8], kB[8];
#pragma unroll
    for (int r = 0; r < 8; ++r) { kA[r] = kOffA(r, g); kB[r] = kOffB(r, g); }

    v8f acc[2][2];
#pragma unroll
    for (int i = 0; i < 2; ++i)
#pragma unroll
        for (int j = 0; j < 2; ++j)
            acc[i][j] = (v8f){0.f, 0.f, 0.f, 0.f, 0.f, 0.f, 0.f, 0.f};

    // Stage one 128x64 A tile + 64x64 B tile into LDS buffer `buf` with
    // direct-to-LDS async DMA (6 instructions per thread => ASYNCcnt += 6).
    auto stageTile = [&](int kt, int buf) {
        unsigned ldsA = (unsigned)(uintptr_t)(&As[buf][0]);
        unsigned ldsB = (unsigned)(uintptr_t)(&Bs[buf][0]);
#pragma unroll
        for (int it = 0; it < 4; ++it) {
            int i = tid + it * 256;
            int row = i >> 3, c8 = i & 7;
            const unsigned short* gp = &A[(m0 + row) * lda + kt + c8 * 8];
            unsigned lp = ldsA + (unsigned)(row * LST + c8 * 8) * 2u;
            asm volatile("global_load_async_to_lds_b128 %0, %1, off"
                         :: "v"(lp), "v"(gp) : "memory");
        }
#pragma unroll
        for (int it = 0; it < 2; ++it) {
            int i = tid + it * 256;
            int row = i >> 3, c8 = i & 7;
            const unsigned short* gp = &B[(n0 + row) * ldb + kt + c8 * 8];
            unsigned lp = ldsB + (unsigned)(row * LST + c8 * 8) * 2u;
            asm volatile("global_load_async_to_lds_b128 %0, %1, off"
                         :: "v"(lp), "v"(gp) : "memory");
        }
    };

    const int nt = K / BK;
    stageTile(0, 0);

    for (int t = 0; t < nt; ++t) {
        if (t + 1 < nt) {
            stageTile((t + 1) * BK, (t + 1) & 1);
            // prev stage (6 async ops) completes once <=6 remain outstanding
            asm volatile("s_wait_asynccnt 0x6" ::: "memory");
        } else {
            asm volatile("s_wait_asynccnt 0x0" ::: "memory");
        }
        __syncthreads();

        const unsigned short* Asb = &As[t & 1][0];
        const unsigned short* Bsb = &Bs[t & 1][0];
#pragma unroll
        for (int ks = 0; ks < BK; ks += 32) {
            Frag a[2], b[2];
#pragma unroll
            for (int ti = 0; ti < 2; ++ti)
#pragma unroll
                for (int r = 0; r < 8; ++r)
                    a[ti].u[r] = *reinterpret_cast<const unsigned*>(
                        &Asb[(wm + ti * 16 + l16) * LST + ks + kA[r]]);
#pragma unroll
            for (int tj = 0; tj < 2; ++tj)
#pragma unroll
                for (int r = 0; r < 8; ++r)
                    b[tj].u[r] = *reinterpret_cast<const unsigned*>(
                        &Bsb[(wn + tj * 16 + l16) * LST + ks + kB[r]]);
#pragma unroll
            for (int ti = 0; ti < 2; ++ti)
#pragma unroll
                for (int tj = 0; tj < 2; ++tj)
                    acc[ti][tj] = wmma_bf16(a[ti].v, b[tj].v, acc[ti][tj]);
        }
        __syncthreads();   // all waves done reading buf before it is re-staged
    }

    // epilogue
#pragma unroll
    for (int ti = 0; ti < 2; ++ti)
#pragma unroll
        for (int tj = 0; tj < 2; ++tj)
#pragma unroll
            for (int r = 0; r < 8; ++r) {
                long m = m0 + wm + ti * 16 + 8 * g + r;
                long n = n0 + wn + tj * 16 + l16;
                float v = acc[ti][tj][r];
                if (bias) v += bias[m];
                if (R)  v += R[bz * sR + m * ldr + n];
                if (C)  C[bz * sC + m * ldc + n] = v;
                if (OB) OB[bz * sO + m * ldo + n] = f2bf(v);
                if (OT) OT[bz * sT + n * ldt + m] = f2bf(v);
            }
}

// ---------------------------------------------------------------------------
// Fused attention: per block = (16 query rows, 1 head, 1 batch), 4 waves.
// Phase 1: S = Q K^T / 8 via WMMA into LDS fp32 [16 x TKV]
// Phase 2: row softmax in LDS, convert to bf16 probs [16 x TKV]
// Phase 3: O = P V via WMMA (V pre-transposed: vT[H][TKV]), bf16 out
// ---------------------------------------------------------------------------
__global__ __launch_bounds__(128)
void attn_kernel(const unsigned short* __restrict__ Q, long sQ,
                 const unsigned short* __restrict__ Km, long sK,
                 const unsigned short* __restrict__ Vt, long sV,
                 unsigned short* __restrict__ O, long sO,
                 int T, int H) {
    extern __shared__ char smem[];
    float*          sc = (float*)smem;                               // [16][T]
    unsigned short* pb = (unsigned short*)(smem + (size_t)16 * T * 4); // [16][T]
    float*          red  = (float*)(smem + (size_t)16 * T * 6);      // [16*8]
    float*          rowm = red + 128;                                 // [16]
    float*          rown = rowm + 16;                                 // [16]

    const int tid  = threadIdx.x;
    const int wave = tid >> 5, lane = tid & 31;
    const int g = lane >> 4, l16 = lane & 15;
    const int m0   = blockIdx.x * 16;
    const int head = blockIdx.y;
    const long bz  = blockIdx.z;
    Q  += bz * sQ;  Km += bz * sK;  Vt += bz * sV;  O += bz * sO;

    int kA[8], kB[8];
#pragma unroll
    for (int r = 0; r < 8; ++r) { kA[r] = kOffA(r, g); kB[r] = kOffB(r, g); }

    // Q fragments (16 x 64, two k-steps of 32)
    Frag aq[2];
#pragma unroll
    for (int s = 0; s < 2; ++s)
#pragma unroll
        for (int r = 0; r < 8; ++r)
            aq[s].u[r] = *reinterpret_cast<const unsigned*>(
                &Q[(long)(m0 + l16) * H + head * 64 + s * 32 + kA[r]]);

    // ---- Phase 1: scores ----
    for (int t0 = wave * 16; t0 < T; t0 += 64) {
        Frag bk[2];
#pragma unroll
        for (int s = 0; s < 2; ++s)
#pragma unroll
            for (int r = 0; r < 8; ++r)
                bk[s].u[r] = *reinterpret_cast<const unsigned*>(
                    &Km[(long)(t0 + l16) * H + head * 64 + s * 32 + kB[r]]);
        v8f a = (v8f){0.f, 0.f, 0.f, 0.f, 0.f, 0.f, 0.f, 0.f};
        a = wmma_bf16(aq[0].v, bk[0].v, a);
        a = wmma_bf16(aq[1].v, bk[1].v, a);
#pragma unroll
        for (int r = 0; r < 8; ++r)
            sc[(8 * g + r) * T + t0 + l16] = a[r] * 0.125f;
    }
    __syncthreads();

    // ---- Phase 2: softmax over T per row ----
    const int row = tid & 15, c8 = tid >> 4;   // 8 threads per row
    float mx = -3.0e38f;
    for (int c = c8; c < T; c += 8) mx = fmaxf(mx, sc[row * T + c]);
    red[row * 8 + c8] = mx;
    __syncthreads();
    if (c8 == 0) {
        float m = red[row * 8];
#pragma unroll
        for (int j = 1; j < 8; ++j) m = fmaxf(m, red[row * 8 + j]);
        rowm[row] = m;
    }
    __syncthreads();
    const float rm = rowm[row];
    float ps = 0.f;
    for (int c = c8; c < T; c += 8) {
        float e = __expf(sc[row * T + c] - rm);
        sc[row * T + c] = e;
        ps += e;
    }
    red[row * 8 + c8] = ps;
    __syncthreads();
    if (c8 == 0) {
        float s = 0.f;
#pragma unroll
        for (int j = 0; j < 8; ++j) s += red[row * 8 + j];
        rown[row] = 1.f / s;
    }
    __syncthreads();
    for (int i = tid; i < 16 * T; i += 128)
        pb[i] = f2bf(sc[i] * rown[i / T]);
    __syncthreads();

    // ---- Phase 3: O = P V ----
    const int d0 = wave * 16;                       // channels within head
    v8f acc = (v8f){0.f, 0.f, 0.f, 0.f, 0.f, 0.f, 0.f, 0.f};
    for (int kt = 0; kt < T; kt += 32) {
        Frag ap, bv;
#pragma unroll
        for (int r = 0; r < 8; ++r)
            ap.u[r] = *reinterpret_cast<const unsigned*>(
                &pb[l16 * T + kt + kA[r]]);
#pragma unroll
        for (int r = 0; r < 8; ++r)
            bv.u[r] = *reinterpret_cast<const unsigned*>(
                &Vt[(long)(head * 64 + d0 + l16) * T + kt + kB[r]]);
        acc = wmma_bf16(ap.v, bv.v, acc);
    }
#pragma unroll
    for (int r = 0; r < 8; ++r)
        O[(long)(m0 + 8 * g + r) * H + head * 64 + d0 + l16] = f2bf(acc[r]);
}

// ---------------------------------------------------------------------------
// Host orchestration
// ---------------------------------------------------------------------------
extern "C" void kernel_launch(void* const* d_in, const int* in_sizes, int n_in,
                              void* d_out, int out_size, void* d_ws, size_t ws_size,
                              hipStream_t stream) {
    const float* hidden = (const float*)d_in[0];   // [2,512,2048]
    const float* W_up   = (const float*)d_in[1];   // [1024,512]
    const float* b_up   = (const float*)d_in[2];   // [1024]
    const float* anw    = (const float*)d_in[3];   // [2048]
    const float* Wq     = (const float*)d_in[4];
    const float* Wk     = (const float*)d_in[5];
    const float* Wv     = (const float*)d_in[6];
    const float* Wo     = (const float*)d_in[7];
    const float* mnw    = (const float*)d_in[8];
    const float* Wg     = (const float*)d_in[9];
    const float* Wu     = (const float*)d_in[10];
    const float* Wd     = (const float*)d_in[11];
    float* out = (float*)d_out;                    // [2,1024,2048]

    char* ws = (char*)d_ws;
    size_t off = 0;
    auto alloc = [&](size_t bytes) -> void* {
        void* p = ws + off;
        off = (off + bytes + 255) & ~(size_t)255;
        return p;
    };
    typedef unsigned short us;

    us* wUpB = (us*)alloc((size_t)SUP * SEQ * 2);
    us* wQB  = (us*)alloc((size_t)HID * HID * 2);
    us* wKB  = (us*)alloc((size_t)HID * HID * 2);
    us* wVB  = (us*)alloc((size_t)HID * HID * 2);
    us* wOB  = (us*)alloc((size_t)HID * HID * 2);
    us* wGB  = (us*)alloc((size_t)INTER * HID * 2);
    us* wUB  = (us*)alloc((size_t)INTER * HID * 2);
    us* wDB  = (us*)alloc((size_t)HID * INTER * 2);
    us* hT   = (us*)alloc((size_t)BATCH * HID * SEQ * 2);
    us* resid= (us*)alloc((size_t)BATCH * TKV * HID * 2);   // [h rows | up rows]
    us* xN   = (us*)alloc((size_t)BATCH * SUP * HID * 2);
    us* qB   = (us*)alloc((size_t)BATCH * SUP * HID * 2);
    us* kB   = (us*)alloc((size_t)BATCH * TKV * HID * 2);
    us* vT   = (us*)alloc((size_t)BATCH * HID * TKV * 2);
    us* oB   = (us*)alloc((size_t)BATCH * SUP * HID * 2);
    us* x2nB = (us*)alloc((size_t)BATCH * SUP * HID * 2);
    us* actB = (us*)alloc((size_t)BATCH * SUP * INTER * 2);
    float* upF  = (float*)alloc((size_t)BATCH * SUP * HID * 4);
    float* xF   = (float*)alloc((size_t)BATCH * SUP * HID * 4);
    float* x2F  = (float*)alloc((size_t)BATCH * SUP * HID * 4);
    float* x2nF = (float*)alloc((size_t)BATCH * SUP * HID * 4);
    float* gF   = (float*)alloc((size_t)BATCH * SUP * INTER * 4);
    float* uF   = (float*)alloc((size_t)BATCH * SUP * INTER * 4);

    auto cvt = [&](const float* in, us* o2, long n) {
        cvt_kernel<<<(unsigned)((n + 255) / 256), 256, 0, stream>>>(in, o2, n);
    };
    // weight conversions (deterministic per call)
    cvt(W_up, wUpB, (long)SUP * SEQ);
    cvt(Wq, wQB, (long)HID * HID);
    cvt(Wk, wKB, (long)HID * HID);
    cvt(Wv, wVB, (long)HID * HID);
    cvt(Wo, wOB, (long)HID * HID);
    cvt(Wg, wGB, (long)INTER * HID);
    cvt(Wu, wUB, (long)INTER * HID);
    cvt(Wd, wDB, (long)HID * INTER);

    // hidden -> bf16 into residual rows [0,512)
    {
        long nPer = (long)SEQ * HID;
        dim3 grid((unsigned)((nPer + 255) / 256), BATCH);
        cvt_batched_kernel<<<grid, 256, 0, stream>>>(
            hidden, nPer, resid, (long)TKV * HID, nPer);
    }
    // hidden -> hT bf16 [H x S] per batch
    {
        dim3 grid(HID / 32, SEQ / 32, BATCH);
        transpose_cvt_kernel<<<grid, dim3(32, 8), 0, stream>>>(
            hidden, hT, SEQ, HID, (long)SEQ * HID, (long)HID * SEQ);
    }

    auto gemm = [&](const us* A, long lda, long sA,
                    const us* Bp, long ldb, long sB,
                    float* C, long ldc, long sC,
                    const float* R, long ldr, long sR,
                    const float* bias,
                    us* Ob, long ldo, long sO,
                    us* Ot, long ldt, long sT,
                    int M, int N, int K) {
        dim3 grid((unsigned)(N / 64), (unsigned)(M / 128), BATCH);
        gemm_bf16_kernel<<<grid, 256, 0, stream>>>(
            A, lda, sA, Bp, ldb, sB, C, ldc, sC, R, ldr, sR,
            bias, Ob, ldo, sO, Ot, ldt, sT, K);
    };

    // up = W_up @ h  (+ b_up): fp32 -> upF, bf16 -> residual rows [512,1536)
    gemm(wUpB, SEQ, 0, hT, SEQ, (long)HID * SEQ,
         upF, HID, (long)SUP * HID, nullptr, 0, 0, b_up,
         resid + (long)SEQ * HID, HID, (long)TKV * HID,
         nullptr, 0, 0, SUP, HID, SEQ);

    // x = rmsnorm(up): fp32 -> xF, bf16 -> xN
    rmsnorm_kernel<<<BATCH * SUP, 256, 0, stream>>>(upF, anw, xF, xN, HID);

    // q/k/v projections
    gemm(xN, HID, (long)SUP * HID, wQB, HID, 0,
         nullptr, 0, 0, nullptr, 0, 0, nullptr,
         qB, HID, (long)SUP * HID, nullptr, 0, 0, SUP, HID, HID);
    gemm(resid, HID, (long)TKV * HID, wKB, HID, 0,
         nullptr, 0, 0, nullptr, 0, 0, nullptr,
         kB, HID, (long)TKV * HID, nullptr, 0, 0, TKV, HID, HID);
    gemm(resid, HID, (long)TKV * HID, wVB, HID, 0,
         nullptr, 0, 0, nullptr, 0, 0, nullptr,
         nullptr, 0, 0, vT, TKV, (long)HID * TKV, TKV, HID, HID);

    // fused attention -> oB (bf16)
    {
        size_t smemB = (size_t)16 * TKV * 6 + (128 + 16 + 16) * 4;
        hipFuncSetAttribute(reinterpret_cast<const void*>(attn_kernel),
                            hipFuncAttributeMaxDynamicSharedMemorySize,
                            (int)smemB);
        dim3 grid(SUP / 16, NHEAD, BATCH);
        attn_kernel<<<grid, 128, smemB, stream>>>(
            qB, (long)SUP * HID, kB, (long)TKV * HID,
            vT, (long)HID * TKV, oB, (long)SUP * HID, TKV, HID);
    }

    // x2 = o @ Wo^T + x
    gemm(oB, HID, (long)SUP * HID, wOB, HID, 0,
         x2F, HID, (long)SUP * HID, xF, HID, (long)SUP * HID, nullptr,
         nullptr, 0, 0, nullptr, 0, 0, SUP, HID, HID);

    // x2n = rmsnorm(x2)
    rmsnorm_kernel<<<BATCH * SUP, 256, 0, stream>>>(x2F, mnw, x2nF, x2nB, HID);

    // gated MLP
    gemm(x2nB, HID, (long)SUP * HID, wGB, HID, 0,
         gF, INTER, (long)SUP * INTER, nullptr, 0, 0, nullptr,
         nullptr, 0, 0, nullptr, 0, 0, SUP, INTER, HID);
    gemm(x2nB, HID, (long)SUP * HID, wUB, HID, 0,
         uF, INTER, (long)SUP * INTER, nullptr, 0, 0, nullptr,
         nullptr, 0, 0, nullptr, 0, 0, SUP, INTER, HID);
    {
        long n = (long)BATCH * SUP * INTER;
        silu_mul_kernel<<<(unsigned)((n + 255) / 256), 256, 0, stream>>>(
            gF, uF, actB, n);
    }
    // out = act @ Wd^T + x2n
    gemm(actB, INTER, (long)SUP * INTER, wDB, INTER, 0,
         out, HID, (long)SUP * HID, x2nF, HID, (long)SUP * HID, nullptr,
         nullptr, 0, 0, nullptr, 0, 0, SUP, HID, INTER);
}
